// MultiHeadAttention_78812649881803
// MI455X (gfx1250) — matrix operand order repaired and checked by reading, not compiled, not run on hardware
//
#include <hip/hip_runtime.h>
#include <hip/hip_bf16.h>
#include <stdint.h>

typedef __attribute__((ext_vector_type(16))) __bf16 v16bf;
typedef __attribute__((ext_vector_type(8)))  float  v8f;

#define EMBED  1024
#define SEQ    2048
#define NBATCH 4
#define NHEAD  16
#define DHEAD  64
#define LOG2E  1.44269504088896340f

// ---------------- WMMA fragment helpers (wave32, gfx1250 layouts) ----------
// A fragment (16x32 bf16): lane m=lane&15, g=lane>>4 holds row m,
//   K = 8g+{0..7} (VGPR0-3) and 16+8g+{0..7} (VGPR4-7).
// p0 points at element (row m, col 8g) of a row-major [*, K] source.
static __device__ __forceinline__ v16bf frag_a(const __bf16* p0) {
  union { uint4 q[2]; v16bf v; } u;
  u.q[0] = *(const uint4*)(p0);
  u.q[1] = *(const uint4*)(p0 + 16);
  return u.v;
}
// B fragment (32x16 bf16): lane n=lane&15, g=lane>>4 holds col n,
//   K = 8g+{0..7} and 16+8g+{0..7}. Source stored n-major; p0 points at (col n, k=8g).
static __device__ __forceinline__ v16bf frag_b(const __bf16* p0) {
  union { uint4 q[2]; v16bf v; } u;
  u.q[0] = *(const uint4*)(p0);
  u.q[1] = *(const uint4*)(p0 + 16);
  return u.v;
}

static __device__ __forceinline__ v8f wmma_bf16(v16bf a, v16bf b, v8f c) {
  return __builtin_amdgcn_wmma_f32_16x16x32_bf16(false, a, false, b,
                                                 (short)0, c, false, false);
}

// ---------------- f32 -> bf16 convert ---------------------------------------
__global__ __launch_bounds__(256) void cvt_f32_bf16(
    const float* __restrict__ in, __bf16* __restrict__ out, int n4) {
  int i = blockIdx.x * 256 + threadIdx.x;
  if (i >= n4) return;
  float4 f = ((const float4*)in)[i];
  union { __bf16 e[4]; uint2 u; } r;
  r.e[0] = (__bf16)f.x; r.e[1] = (__bf16)f.y;
  r.e[2] = (__bf16)f.z; r.e[3] = (__bf16)f.w;
  ((uint2*)out)[i] = r.u;
}

// ---------------- f32 [K][N] -> bf16 transposed [N][K] ----------------------
// 32x32 tile via LDS, 256 threads (32x8).
__global__ __launch_bounds__(256) void transpose_f32_bf16(
    const float* __restrict__ in, __bf16* __restrict__ out, int K, int N) {
  __shared__ float t[32][33];
  const int k0 = blockIdx.y * 32, n0 = blockIdx.x * 32;
  const int tx = threadIdx.x & 31, ty = threadIdx.x >> 5;
#pragma unroll
  for (int p = 0; p < 4; ++p)
    t[ty + p * 8][tx] = in[(size_t)(k0 + ty + p * 8) * N + n0 + tx];
  __syncthreads();
#pragma unroll
  for (int p = 0; p < 4; ++p)
    out[(size_t)(n0 + ty + p * 8) * K + k0 + tx] = (__bf16)t[tx][ty + p * 8];
}

// ---------------- LDS-free streaming bf16 GEMM ------------------------------
// C[M,N] = A[M,K] * Bt[N,K]^T + bias.  Block = 8 waves (2 M x 4 N) covering a
// 128x256 tile; each wave owns a 64x64 tile (4x4 WMMA accumulators) and
// streams A/B fragments directly from global (L2-resident), no barriers.
// Per K=32 step: 8 fragment loads (8 KB) feed 16 WMMAs (262 KFLOP).
template <typename OutT>
__global__ __launch_bounds__(256) void gemm_bf16_wmma(
    const __bf16* __restrict__ A,   // [M,K] row-major
    const __bf16* __restrict__ Bt,  // [N,K] row-major (pre-transposed weights)
    const float* __restrict__ bias, OutT* __restrict__ C,
    int M, int N, int K) {
  const int lane = threadIdx.x & 31;
  const int wave = threadIdx.x >> 5;
  const int m0 = blockIdx.y * 128 + (wave >> 2) * 64;
  const int n0 = blockIdx.x * 256 + (wave & 3) * 64;
  const int lm = lane & 15;
  const int g  = lane >> 4;

  const __bf16* ap[4];
  const __bf16* bp[4];
#pragma unroll
  for (int i = 0; i < 4; ++i)
    ap[i] = A + (size_t)(m0 + i * 16 + lm) * K + 8 * g;
#pragma unroll
  for (int j = 0; j < 4; ++j)
    bp[j] = Bt + (size_t)(n0 + j * 16 + lm) * K + 8 * g;

  v8f acc[4][4] = {};

  for (int kb = 0; kb < K; kb += 32) {
    if ((kb & 32) == 0) {  // one 128B cacheline ahead, every other step
#pragma unroll
      for (int i = 0; i < 4; ++i) __builtin_prefetch(ap[i] + kb + 64, 0, 1);
#pragma unroll
      for (int j = 0; j < 4; ++j) __builtin_prefetch(bp[j] + kb + 64, 0, 1);
    }
    v16bf af[4], bf[4];
#pragma unroll
    for (int i = 0; i < 4; ++i) af[i] = frag_a(ap[i] + kb);
#pragma unroll
    for (int j = 0; j < 4; ++j) bf[j] = frag_b(bp[j] + kb);
#pragma unroll
    for (int i = 0; i < 4; ++i)
#pragma unroll
      for (int j = 0; j < 4; ++j)
        acc[i][j] = wmma_bf16(af[i], bf[j], acc[i][j]);
  }

#pragma unroll
  for (int i = 0; i < 4; ++i) {
#pragma unroll
    for (int j = 0; j < 4; ++j) {
      int gm = m0 + i * 16 + 8 * g;  // + r below
      int gn = n0 + j * 16 + lm;
      float bv = bias[gn];
#pragma unroll
      for (int r = 0; r < 8; ++r)
        C[(size_t)(gm + r) * N + gn] = (OutT)(acc[i][j][r] + bv);
    }
  }
}

// ---------------- flash attention ------------------------------------------
// Grid: (B*H, S/128). 8 waves; wave w owns queries [q0, q0+16).
// qkv layout: [(b*S+s)][t*1024 + h*64 + d] bf16.
#define VT 136  // LDS row stride (128+8) bf16 -> 272B, 16B aligned

__global__ __launch_bounds__(256) void attn_wmma(
    const __bf16* __restrict__ qkv, __bf16* __restrict__ out) {
  __shared__ __align__(16) __bf16 Vt[DHEAD * VT];   // [d][key]
  __shared__ __align__(16) __bf16 Ps[8 * 16 * VT];  // per-wave P scratch

  const int tid  = threadIdx.x;
  const int lane = tid & 31;
  const int wave = tid >> 5;
  const int lm = lane & 15;
  const int g  = lane >> 4;
  const int b  = blockIdx.x >> 4;
  const int h  = blockIdx.x & 15;
  const int q0 = blockIdx.y * 128 + wave * 16;

  // Q A-fragments straight from global (contiguous along D per lane)
  v16bf qf[2];
  {
    const __bf16* p =
        qkv + (size_t)(b * SEQ + q0 + lm) * (3 * EMBED) + h * DHEAD + 8 * g;
    qf[0] = frag_a(p);
    qf[1] = frag_a(p + 32);
  }

  float m_i[8], l_i[8];
#pragma unroll
  for (int r = 0; r < 8; ++r) { m_i[r] = -1e30f; l_i[r] = 0.0f; }
  v8f o[4] = {};

  __bf16* pw = Ps + wave * 16 * VT;

  for (int kv = 0; kv < SEQ; kv += 128) {
    __syncthreads();
    // stage V tile transposed into LDS: Vt[d][key]
    {
      int key = tid >> 3;        // 0..31
      int d0  = (tid & 7) * 8;
#pragma unroll
      for (int p = 0; p < 4; ++p) {
        union { uint4 q; __bf16 e[8]; } u;
        u.q = *(const uint4*)(qkv +
              (size_t)(b * SEQ + kv + p * 32 + key) * (3 * EMBED) +
              2 * EMBED + h * DHEAD + d0);
#pragma unroll
        for (int e = 0; e < 8; ++e)
          Vt[(d0 + e) * VT + p * 32 + key] = u.e[e];
      }
    }
    __syncthreads();

    // scores S = Q*K^T : 8 tiles of 16 keys; K B-fragments from global
    v8f s[8];
#pragma unroll
    for (int nt = 0; nt < 8; ++nt) {
      const __bf16* kp =
          qkv + (size_t)(b * SEQ + kv + nt * 16 + lm) * (3 * EMBED) +
          EMBED + h * DHEAD + 8 * g;
      v8f a = {};
      a = wmma_bf16(qf[0], frag_b(kp), a);
      a = wmma_bf16(qf[1], frag_b(kp + 32), a);
      s[nt] = a;
    }
#pragma unroll
    for (int nt = 0; nt < 8; ++nt)
#pragma unroll
      for (int r = 0; r < 8; ++r) s[nt][r] *= 0.125f;  // 1/sqrt(64)

    // row max: combine 8 tiles, then 16-lane half-wave reduce
    float rmax[8];
#pragma unroll
    for (int r = 0; r < 8; ++r) {
      float v = s[0][r];
#pragma unroll
      for (int nt = 1; nt < 8; ++nt) v = fmaxf(v, s[nt][r]);
      rmax[r] = v;
    }
#pragma unroll
    for (int off = 8; off >= 1; off >>= 1)
#pragma unroll
      for (int r = 0; r < 8; ++r)
        rmax[r] = fmaxf(rmax[r], __shfl_xor(rmax[r], off, 32));

    float alpha[8];
#pragma unroll
    for (int r = 0; r < 8; ++r) {
      float mn = fmaxf(m_i[r], rmax[r]);
      alpha[r] = exp2f((m_i[r] - mn) * LOG2E);
      m_i[r] = mn;
    }
#pragma unroll
    for (int j = 0; j < 4; ++j)
#pragma unroll
      for (int r = 0; r < 8; ++r) o[j][r] *= alpha[r];

    float rsum[8];
#pragma unroll
    for (int r = 0; r < 8; ++r) rsum[r] = 0.0f;
#pragma unroll
    for (int nt = 0; nt < 8; ++nt)
#pragma unroll
      for (int r = 0; r < 8; ++r) {
        float e = exp2f((s[nt][r] - m_i[r]) * LOG2E);
        s[nt][r] = e;
        rsum[r] += e;
      }
#pragma unroll
    for (int off = 8; off >= 1; off >>= 1)
#pragma unroll
      for (int r = 0; r < 8; ++r)
        rsum[r] += __shfl_xor(rsum[r], off, 32);
#pragma unroll
    for (int r = 0; r < 8; ++r) l_i[r] = l_i[r] * alpha[r] + rsum[r];

    // P (C layout) -> LDS row-major bf16 so it can be re-read as A fragments
#pragma unroll
    for (int nt = 0; nt < 8; ++nt)
#pragma unroll
      for (int r = 0; r < 8; ++r)
        pw[(8 * g + r) * VT + nt * 16 + lm] = (__bf16)s[nt][r];

    // O += P * V  (K loop over 128 keys in 4 steps of 32)
#pragma unroll
    for (int ks = 0; ks < 4; ++ks) {
      v16bf pf = frag_a(pw + lm * VT + ks * 32 + 8 * g);
#pragma unroll
      for (int j = 0; j < 4; ++j) {
        v16bf vf = frag_b(Vt + (j * 16 + lm) * VT + ks * 32 + 8 * g);
        o[j] = wmma_bf16(pf, vf, o[j]);
      }
    }
  }

  // normalize and store bf16 [B*S, EMBED] with col = h*64 + d
#pragma unroll
  for (int j = 0; j < 4; ++j)
#pragma unroll
    for (int r = 0; r < 8; ++r) {
      float v = o[j][r] / l_i[r];
      out[(size_t)(b * SEQ + q0 + 8 * g + r) * EMBED + h * DHEAD + j * 16 + lm] =
          (__bf16)v;
    }
}

// ---------------- launch -----------------------------------------------------
extern "C" void kernel_launch(void* const* d_in, const int* in_sizes, int n_in,
                              void* d_out, int out_size, void* d_ws, size_t ws_size,
                              hipStream_t stream) {
  (void)in_sizes; (void)n_in; (void)out_size; (void)ws_size;
  const float* x    = (const float*)d_in[0];
  const float* Wqkv = (const float*)d_in[1];
  const float* bqkv = (const float*)d_in[2];
  const float* Wo   = (const float*)d_in[3];
  const float* bo   = (const float*)d_in[4];
  float* out = (float*)d_out;

  const size_t nX    = (size_t)NBATCH * SEQ * EMBED;      // 8388608
  const size_t nWqkv = (size_t)EMBED * 3 * EMBED;         // 3145728
  const size_t nWo   = (size_t)EMBED * EMBED;             // 1048576
  const size_t nQKV  = (size_t)NBATCH * SEQ * 3 * EMBED;  // 25165824

  char* w = (char*)d_ws;
  auto carve = [&](size_t bytes) {
    char* p = w;
    w += (bytes + 255) & ~(size_t)255;
    return p;
  };
  __bf16* xb    = (__bf16*)carve(nX * 2);
  __bf16* wqkvt = (__bf16*)carve(nWqkv * 2);  // [3E][E] transposed
  __bf16* wot   = (__bf16*)carve(nWo * 2);    // [E][E] transposed
  __bf16* qkvb  = (__bf16*)carve(nQKV * 2);
  __bf16* attnb = (__bf16*)carve(nX * 2);

  cvt_f32_bf16<<<(int)(nX / 4 / 256), 256, 0, stream>>>(x, xb, (int)(nX / 4));
  transpose_f32_bf16<<<dim3(3 * EMBED / 32, EMBED / 32), 256, 0, stream>>>(
      Wqkv, wqkvt, EMBED, 3 * EMBED);
  transpose_f32_bf16<<<dim3(EMBED / 32, EMBED / 32), 256, 0, stream>>>(
      Wo, wot, EMBED, EMBED);

  // qkv = x @ Wqkv + bqkv   (M=8192, N=3072, K=1024) -> bf16
  gemm_bf16_wmma<__bf16><<<dim3(3 * EMBED / 256, NBATCH * SEQ / 128), 256, 0, stream>>>(
      xb, wqkvt, bqkv, qkvb, NBATCH * SEQ, 3 * EMBED, EMBED);

  // flash attention -> bf16 [B*S, EMBED]
  attn_wmma<<<dim3(NBATCH * NHEAD, SEQ / 128), 256, 0, stream>>>(qkvb, attnb);

  // out = attn @ Wo + bo    (M=8192, N=1024, K=1024) -> f32
  gemm_bf16_wmma<float><<<dim3(EMBED / 256, NBATCH * SEQ / 128), 256, 0, stream>>>(
      attnb, wot, bo, out, NBATCH * SEQ, EMBED, EMBED);
}
